// FusedMoE_15401752723974
// MI455X (gfx1250) — matrix operand order repaired and checked by reading, not compiled, not run on hardware
//
#include <hip/hip_runtime.h>
#include <hip/hip_bf16.h>
#include <math.h>

// Problem constants (match the reference).
#define T_TOK 2048
#define H_DIM 2048
#define I_DIM 1408
#define E_NUM 8
#define TOPKN 2
#define TI_DIM (2 * I_DIM)          // 2816
#define P_PAIR (T_TOK * TOPKN)      // 4096

typedef __bf16 bf16_t;
typedef __attribute__((ext_vector_type(8)))  bf16_t v8bf;
typedef __attribute__((ext_vector_type(16))) bf16_t v16bf;
typedef __attribute__((ext_vector_type(8)))  float  v8f;
typedef __attribute__((ext_vector_type(4)))  float  v4f;

// ---------------- fp32 -> bf16 (round to nearest even) ----------------
__device__ __forceinline__ bf16_t f2bf(float f) {
    union { float f; unsigned u; } v; v.f = f;
    unsigned u = v.u;
    unsigned r = u + 0x7FFFu + ((u >> 16) & 1u);
    unsigned short h = (unsigned short)(r >> 16);
    bf16_t b;
    __builtin_memcpy(&b, &h, 2);
    return b;
}

__global__ void cvt_f32_to_bf16(const float* __restrict__ in,
                                bf16_t* __restrict__ out, int n) {
    int i = (blockIdx.x * blockDim.x + threadIdx.x) * 8;
    if (i >= n) return;
    v4f a = *(const v4f*)(in + i);
    v4f b = *(const v4f*)(in + i + 4);
    v8bf o;
    o[0] = f2bf(a[0]); o[1] = f2bf(a[1]); o[2] = f2bf(a[2]); o[3] = f2bf(a[3]);
    o[4] = f2bf(b[0]); o[5] = f2bf(b[1]); o[6] = f2bf(b[2]); o[7] = f2bf(b[3]);
    *(v8bf*)(out + i) = o;
}

// ---------------- router: top-2 of 8 with renormalized softmax ----------------
__global__ void router_kernel(const float* __restrict__ logits,
                              int* __restrict__ eid, float* __restrict__ wts) {
    int t = blockIdx.x * blockDim.x + threadIdx.x;
    if (t >= T_TOK) return;
    const float* l = logits + (size_t)t * E_NUM;
    float v1 = -INFINITY, v2 = -INFINITY;
    int i1 = 0, i2 = 0;
#pragma unroll
    for (int i = 0; i < E_NUM; ++i) {
        float x = l[i];
        if (x > v1) { v2 = v1; i2 = i1; v1 = x; i1 = i; }
        else if (x > v2) { v2 = x; i2 = i; }
    }
    // softmax over all experts then renormalize over top-2 == softmax over {v1,v2}
    float r = __expf(v2 - v1);
    float inv = 1.0f / (1.0f + r);
    eid[t * 2 + 0] = i1;  wts[t * 2 + 0] = inv;
    eid[t * 2 + 1] = i2;  wts[t * 2 + 1] = r * inv;
}

// ---------------- deterministic stable grouping (counting scan) ----------------
__global__ void scatter_kernel(const int* __restrict__ eid, const float* __restrict__ wts,
                               int* __restrict__ ptok, float* __restrict__ pw,
                               int* __restrict__ ebase, int* __restrict__ ecnt) {
    __shared__ int se[P_PAIR];
    for (int i = threadIdx.x; i < P_PAIR; i += blockDim.x) se[i] = eid[i];
    __syncthreads();
    int p = blockIdx.x * blockDim.x + threadIdx.x;
    if (p < P_PAIR) {
        int e = se[p];
        int slot = 0;
        for (int q = 0; q < P_PAIR; ++q) {
            int eq = se[q];
            slot += (eq < e) | ((eq == e) & (q < p));
        }
        ptok[slot] = p >> 1;     // token id
        pw[slot]   = wts[p];     // combine weight
    }
    if (p < E_NUM) {
        int c = 0, b = 0;
        for (int q = 0; q < P_PAIR; ++q) { int eq = se[q]; c += (eq == p); b += (eq < p); }
        ecnt[p] = c; ebase[p] = b;
    }
}

// ---------------- zero out ----------------
__global__ void zero_kernel(float* __restrict__ out, int n) {
    int i = (blockIdx.x * blockDim.x + threadIdx.x) * 4;
    if (i < n) *(v4f*)(out + i) = (v4f){0.f, 0.f, 0.f, 0.f};
}

// ---------------- WMMA fragment helpers ----------------
// A fragment (16x32 bf16, M x K): lane r=lane&15 is row; lane group g=lane>>4
// holds contiguous K chunk [g*16, g*16+16).
__device__ __forceinline__ v16bf ldA(const bf16_t* __restrict__ p) {
    v8bf c0 = *(const v8bf*)p;
    v8bf c1 = *(const v8bf*)(p + 8);
    return __builtin_shufflevector(c0, c1, 0,1,2,3,4,5,6,7,8,9,10,11,12,13,14,15);
}
// B fragment (32x16 bf16, K x N) loaded from N x K row-major weights:
// lane r=lane&15 is column n; lane group g holds contiguous K chunk [g*16, g*16+16).
__device__ __forceinline__ v16bf ldB(const bf16_t* __restrict__ p) {
    v8bf c0 = *(const v8bf*)p;
    v8bf c1 = *(const v8bf*)(p + 8);
    return __builtin_shufflevector(c0, c1, 0,1,2,3,4,5,6,7,8,9,10,11,12,13,14,15);
}
__device__ __forceinline__ v8f wmma_bf16(v16bf a, v16bf b, v8f c) {
    return __builtin_amdgcn_wmma_f32_16x16x32_bf16(false, a, false, b, (short)0, c,
                                                   false, false);
}

// ---------------- GEMM1: act = silu(x@w13_gate^T) * (x@w13_up^T) ----------------
// grid: (I/64, T/128, E), block: 256 (8 waves). Block tile: 128 rows x 64 inter cols.
// Wave (wr=wv>>1, wc=wv&1): rows [wr*32, wr*32+32), inter cols [wc*32, wc*32+32),
// computing both gate (col c) and up (col c + I) accumulators.
__global__ __launch_bounds__(256)
void gemm1_kernel(const bf16_t* __restrict__ xb, const bf16_t* __restrict__ w13b,
                  const int* __restrict__ ptok, const int* __restrict__ ebase,
                  const int* __restrict__ ecnt, bf16_t* __restrict__ act) {
    const int e   = blockIdx.z;
    const int n_e = ecnt[e];
    const int m0  = blockIdx.y * 128;
    if (m0 >= n_e) return;
    const int base = ebase[e];

    const int lane = threadIdx.x & 31;
    const int wv   = threadIdx.x >> 5;
    const int wr   = wv >> 1;
    const int wc   = wv & 1;
    const int r    = lane & 15;
    const int hi   = lane >> 4;
    const int kbA  = hi * 16;   // A chunk base
    const int kbB  = hi * 16;   // B chunk base

    const int mw = m0 + wr * 32;

    // A row pointers (gather token rows; clamp padding rows to a valid row)
    const bf16_t* pa[2];
#pragma unroll
    for (int f = 0; f < 2; ++f) {
        int rp = mw + f * 16 + r;
        int rc = rp < n_e ? rp : (n_e - 1);
        int tok = ptok[base + rc];
        pa[f] = xb + (size_t)tok * H_DIM + kbA;
    }
    // B row pointers: gate cols and matching up cols
    const int c0 = blockIdx.x * 64 + wc * 32;
    const bf16_t* wbase = w13b + (size_t)e * TI_DIM * H_DIM;
    const bf16_t* pb[4];
#pragma unroll
    for (int f = 0; f < 2; ++f) {
        int gc = c0 + f * 16 + r;
        pb[f]     = wbase + (size_t)gc * H_DIM + kbB;             // gate
        pb[2 + f] = wbase + (size_t)(gc + I_DIM) * H_DIM + kbB;   // up
    }

    v8f acc[2][4];
#pragma unroll
    for (int f = 0; f < 2; ++f)
#pragma unroll
        for (int j = 0; j < 4; ++j) acc[f][j] = (v8f)(0.f);

#pragma unroll 2
    for (int k = 0; k < H_DIM; k += 32) {
        if (k + 256 <= H_DIM) {
            __builtin_prefetch(pb[0] + k + 256, 0, 1);
            __builtin_prefetch(pb[2] + k + 256, 0, 1);
        }
        v16bf a0 = ldA(pa[0] + k);
        v16bf a1 = ldA(pa[1] + k);
        v16bf b0 = ldB(pb[0] + k);
        v16bf b1 = ldB(pb[1] + k);
        v16bf b2 = ldB(pb[2] + k);
        v16bf b3 = ldB(pb[3] + k);
        acc[0][0] = wmma_bf16(a0, b0, acc[0][0]);
        acc[0][1] = wmma_bf16(a0, b1, acc[0][1]);
        acc[0][2] = wmma_bf16(a0, b2, acc[0][2]);
        acc[0][3] = wmma_bf16(a0, b3, acc[0][3]);
        acc[1][0] = wmma_bf16(a1, b0, acc[1][0]);
        acc[1][1] = wmma_bf16(a1, b1, acc[1][1]);
        acc[1][2] = wmma_bf16(a1, b2, acc[1][2]);
        acc[1][3] = wmma_bf16(a1, b3, acc[1][3]);
    }

    // Epilogue: silu(gate)*up -> bf16 act rows (compact pair-slot indexing)
#pragma unroll
    for (int f = 0; f < 2; ++f) {
#pragma unroll
        for (int j = 0; j < 2; ++j) {
            int c = c0 + j * 16 + r;
#pragma unroll
            for (int v = 0; v < 8; ++v) {
                int rp = mw + f * 16 + hi * 8 + v;
                if (rp < n_e) {
                    float g = acc[f][j][v];
                    float u = acc[f][2 + j][v];
                    float s = g / (1.0f + __expf(-g));
                    act[(size_t)(base + rp) * I_DIM + c] = f2bf(s * u);
                }
            }
        }
    }
}

// ---------------- GEMM2: out[t] += w * (act @ w2^T) ----------------
// grid: (H/128, T/128, E), block: 256 (8 waves). Wave tile: 32 rows x 64 cols.
__global__ __launch_bounds__(256)
void gemm2_kernel(const bf16_t* __restrict__ act, const bf16_t* __restrict__ w2b,
                  const int* __restrict__ ptok, const float* __restrict__ pw,
                  const int* __restrict__ ebase, const int* __restrict__ ecnt,
                  float* __restrict__ out) {
    const int e   = blockIdx.z;
    const int n_e = ecnt[e];
    const int m0  = blockIdx.y * 128;
    if (m0 >= n_e) return;
    const int base = ebase[e];

    const int lane = threadIdx.x & 31;
    const int wv   = threadIdx.x >> 5;
    const int wr   = wv >> 1;
    const int wc   = wv & 1;
    const int r    = lane & 15;
    const int hi   = lane >> 4;
    const int kbA  = hi * 16;
    const int kbB  = hi * 16;

    const int mw = m0 + wr * 32;

    const bf16_t* pa[2];
#pragma unroll
    for (int f = 0; f < 2; ++f) {
        int rp = mw + f * 16 + r;
        int rc = rp < n_e ? rp : (n_e - 1);
        pa[f] = act + (size_t)(base + rc) * I_DIM + kbA;
    }
    const int nb = blockIdx.x * 128 + wc * 64;
    const bf16_t* wbase = w2b + (size_t)e * H_DIM * I_DIM;
    const bf16_t* pb[4];
#pragma unroll
    for (int j = 0; j < 4; ++j)
        pb[j] = wbase + (size_t)(nb + j * 16 + r) * I_DIM + kbB;

    v8f acc[2][4];
#pragma unroll
    for (int f = 0; f < 2; ++f)
#pragma unroll
        for (int j = 0; j < 4; ++j) acc[f][j] = (v8f)(0.f);

#pragma unroll 2
    for (int k = 0; k < I_DIM; k += 32) {
        if (k + 256 <= I_DIM) {
            __builtin_prefetch(pb[0] + k + 256, 0, 1);
            __builtin_prefetch(pb[2] + k + 256, 0, 1);
        }
        v16bf a0 = ldA(pa[0] + k);
        v16bf a1 = ldA(pa[1] + k);
        v16bf b0 = ldB(pb[0] + k);
        v16bf b1 = ldB(pb[1] + k);
        v16bf b2 = ldB(pb[2] + k);
        v16bf b3 = ldB(pb[3] + k);
        acc[0][0] = wmma_bf16(a0, b0, acc[0][0]);
        acc[0][1] = wmma_bf16(a0, b1, acc[0][1]);
        acc[0][2] = wmma_bf16(a0, b2, acc[0][2]);
        acc[0][3] = wmma_bf16(a0, b3, acc[0][3]);
        acc[1][0] = wmma_bf16(a1, b0, acc[1][0]);
        acc[1][1] = wmma_bf16(a1, b1, acc[1][1]);
        acc[1][2] = wmma_bf16(a1, b2, acc[1][2]);
        acc[1][3] = wmma_bf16(a1, b3, acc[1][3]);
    }

    // Per-lane row metadata (token id + combine weight) for the 16 rows this lane owns
    int   trow[2][8];
    float twt[2][8];
#pragma unroll
    for (int f = 0; f < 2; ++f)
#pragma unroll
        for (int v = 0; v < 8; ++v) {
            int rp = mw + f * 16 + hi * 8 + v;
            if (rp < n_e) { trow[f][v] = ptok[base + rp]; twt[f][v] = pw[base + rp]; }
            else          { trow[f][v] = -1;              twt[f][v] = 0.f; }
        }

#pragma unroll
    for (int f = 0; f < 2; ++f)
#pragma unroll
        for (int j = 0; j < 4; ++j) {
            int c = nb + j * 16 + r;
#pragma unroll
            for (int v = 0; v < 8; ++v) {
                if (trow[f][v] >= 0)
                    atomicAdd(out + (size_t)trow[f][v] * H_DIM + c, twt[f][v] * acc[f][j][v]);
            }
        }
}

// ---------------- host launcher ----------------
extern "C" void kernel_launch(void* const* d_in, const int* in_sizes, int n_in,
                              void* d_out, int out_size, void* d_ws, size_t ws_size,
                              hipStream_t stream) {
    const float* x   = (const float*)d_in[0];
    const float* rl  = (const float*)d_in[1];
    const float* w13 = (const float*)d_in[2];
    const float* w2  = (const float*)d_in[3];
    // d_in[4] = top_k scalar (compile-time TOPKN)
    float* out = (float*)d_out;
    (void)in_sizes; (void)n_in; (void)out_size; (void)ws_size;

    char* ws = (char*)d_ws;
    size_t off = 0;
    auto take = [&](size_t bytes) -> char* {
        char* p = ws + off;
        off += (bytes + 255) & ~(size_t)255;
        return p;
    };
    bf16_t* xb    = (bf16_t*)take((size_t)T_TOK * H_DIM * 2);
    bf16_t* w13b  = (bf16_t*)take((size_t)E_NUM * TI_DIM * H_DIM * 2);
    bf16_t* w2b   = (bf16_t*)take((size_t)E_NUM * H_DIM * I_DIM * 2);
    bf16_t* act   = (bf16_t*)take((size_t)P_PAIR * I_DIM * 2);
    int*    eid   = (int*)take((size_t)P_PAIR * 4);
    float*  wts   = (float*)take((size_t)P_PAIR * 4);
    int*    ptok  = (int*)take((size_t)P_PAIR * 4);
    float*  pw    = (float*)take((size_t)P_PAIR * 4);
    int*    ebase = (int*)take((size_t)E_NUM * 4);
    int*    ecnt  = (int*)take((size_t)E_NUM * 4);

    const int nX   = T_TOK * H_DIM;
    const int nW13 = E_NUM * TI_DIM * H_DIM;
    const int nW2  = E_NUM * H_DIM * I_DIM;

    cvt_f32_to_bf16<<<(nX / 8 + 255) / 256, 256, 0, stream>>>(x, xb, nX);
    cvt_f32_to_bf16<<<(nW13 / 8 + 255) / 256, 256, 0, stream>>>(w13, w13b, nW13);
    cvt_f32_to_bf16<<<(nW2 / 8 + 255) / 256, 256, 0, stream>>>(w2, w2b, nW2);

    router_kernel<<<T_TOK / 256, 256, 0, stream>>>(rl, eid, wts);
    scatter_kernel<<<P_PAIR / 256, 256, 0, stream>>>(eid, wts, ptok, pw, ebase, ecnt);

    zero_kernel<<<(T_TOK * H_DIM / 4 + 255) / 256, 256, 0, stream>>>(out, T_TOK * H_DIM);

    gemm1_kernel<<<dim3(I_DIM / 64, T_TOK / 128, E_NUM), 256, 0, stream>>>(
        xb, w13b, ptok, ebase, ecnt, act);
    gemm2_kernel<<<dim3(H_DIM / 128, T_TOK / 128, E_NUM), 256, 0, stream>>>(
        act, w2b, ptok, pw, ebase, ecnt, out);
}
